// TransformerLSTMEncoder_75574244540518
// MI455X (gfx1250) — compile-verified, matrix operand order checked by reading, not run on hardware
//
#include <hip/hip_runtime.h>
#include <hip/hip_bf16.h>
#include <math.h>

// ---------------------------------------------------------------------------
// TransformerLSTMEncoder for MI455X (gfx1250, wave32, WMMA)
// All GEMMs: f32 in -> bf16 staged in LDS -> v_wmma_f32_16x16x32_bf16 -> f32 out
// ---------------------------------------------------------------------------

typedef __bf16 bf16;
typedef __bf16 v16bf __attribute__((ext_vector_type(16)));
typedef __bf16 v8bf  __attribute__((ext_vector_type(8)));
typedef float  v8f   __attribute__((ext_vector_type(8)));
typedef float  v4f   __attribute__((ext_vector_type(4)));

#define WMMA_BF16(a, b, c) \
  __builtin_amdgcn_wmma_f32_16x16x32_bf16(false, (a), false, (b), (short)0, (c), false, false)

static __device__ __forceinline__ v8bf cvt8(v4f a, v4f b) {
  v8bf t;
  t[0] = (bf16)a[0]; t[1] = (bf16)a[1]; t[2] = (bf16)a[2]; t[3] = (bf16)a[3];
  t[4] = (bf16)b[0]; t[5] = (bf16)b[1]; t[6] = (bf16)b[2]; t[7] = (bf16)b[3];
  return t;
}

// Fragment-ready LDS layout for a 16x32 (A) / 32x16 (B) bf16 operand tile:
//   lane = r + ((k&8)?16:0),  element e = (k&16) ? (k&7)+8 : (k&7)
// so each lane's v16bf operand is one contiguous 32-byte LDS read, and any
// 8-aligned run of k at fixed r is one contiguous v8bf (16-byte) LDS write.

// ---------------------------------------------------------------------------
// Generic batched GEMM: C[z] = act( A[z](MxK) @ B[z](KxN) + bias[z] )
// block = 256 threads (8 waves), tile 64x128, BK = 32
// wave w: mt = w>>1 (16-row subtile), nt = (w&1)*4 .. +4  -> 4 WMMAs per wave
// grid = (N/128, M/64, batch)
// ---------------------------------------------------------------------------
__global__ __launch_bounds__(256) void gemm_bf16_kernel(
    const float* __restrict__ A, long bsA, int lda,
    const float* __restrict__ B, long bsB, int ldb,
    const float* __restrict__ bias, long bsBias,
    float* __restrict__ C, long bsC, int ldc,
    int K, int relu)
{
  extern __shared__ char smem[];
  bf16* ldsA = (bf16*)smem;            // 64x32 bf16 (4KB), fragment layout
  bf16* ldsB = (bf16*)(smem + 4096);   // 32x128 bf16 (8KB), fragment layout

  const int tid  = threadIdx.x;
  const int lane = tid & 31;
  const int w    = tid >> 5;
  const int mt   = w >> 1;          // wave's 16-row subtile (0..3)
  const int nt0  = (w & 1) * 4;     // wave's first 16-col subtile
  const int row0 = blockIdx.y * 64;
  const int col0 = blockIdx.x * 128;
  const long zb  = blockIdx.z;

  const float* Ab = A + zb * bsA;
  const float* Bb = B + zb * bsB;
  float*       Cb = C + zb * bsC;

  v8f acc[4] = {};

  // A staging: thread -> (row sam, 8 consecutive k at sak)
  const int sam = tid >> 2;
  const int sak = (tid & 3) * 8;
  const int aBase = (((sam >> 4) * 32 + (sam & 15) + ((sak & 8) ? 16 : 0)) * 16)
                    + ((sak & 16) ? 8 : 0);
  // B staging: thread -> (col sbn, 16 k at sbk in {0,16})
  const int sbn  = tid & 127;
  const int sbk  = (tid >> 7) * 16;
  const int bnt  = sbn >> 4;
  const int br   = sbn & 15;
  const int bEo  = (sbk & 16) ? 8 : 0;
  const int bBase0 = bnt * 512 + br * 16 + bEo;          // k = sbk..sbk+7   (lanes 0-15 half)
  const int bBase1 = bnt * 512 + (br + 16) * 16 + bEo;   // k = sbk+8..+15  (lanes 16-31 half)

  for (int kk = 0; kk < K; kk += 32) {
    {
      const float* src = Ab + (long)(row0 + sam) * lda + kk + sak;
      v4f f0 = *(const v4f*)src;
      v4f f1 = *(const v4f*)(src + 4);
      *(v8bf*)&ldsA[aBase] = cvt8(f0, f1);
      if (kk + 32 < K) __builtin_prefetch(src + 32, 0, 1);   // global_prefetch_b8
    }
    {
      const float* src = Bb + (long)(kk + sbk) * ldb + col0 + sbn;
      v8bf t0, t1;
      #pragma unroll
      for (int i = 0; i < 8; ++i) t0[i] = (bf16)src[(long)i * ldb];
      #pragma unroll
      for (int i = 0; i < 8; ++i) t1[i] = (bf16)src[(long)(8 + i) * ldb];
      *(v8bf*)&ldsB[bBase0] = t0;
      *(v8bf*)&ldsB[bBase1] = t1;
      if (kk + 32 < K) __builtin_prefetch(src + (long)32 * ldb, 0, 1);
    }
    __syncthreads();
    v16bf aF = *(const v16bf*)&ldsA[(mt * 32 + lane) * 16];
    #pragma unroll
    for (int j = 0; j < 4; ++j) {
      v16bf bF = *(const v16bf*)&ldsB[((nt0 + j) * 32 + lane) * 16];
      acc[j] = WMMA_BF16(aF, bF, acc[j]);
    }
    __syncthreads();
  }

  // epilogue (C/D layout: vgpr i -> M = i + (lane<16?0:8), N = lane&15)
  const int mBase = row0 + mt * 16 + ((lane & 16) ? 8 : 0);
  #pragma unroll
  for (int j = 0; j < 4; ++j) {
    const int col = col0 + (nt0 + j) * 16 + (lane & 15);
    const float bv = bias ? bias[zb * bsBias + col] : 0.0f;
    #pragma unroll
    for (int i = 0; i < 8; ++i) {
      float y = acc[j][i] + bv;
      if (relu) y = y > 0.0f ? y : 0.0f;
      Cb[(long)(mBase + i) * ldc + col] = y;
    }
  }
}

// ---------------------------------------------------------------------------
// Attention: one workgroup = (64-query block) x (b, head).
// q/k/v layout: [B, T, NH*DH] with column = n*64 + d.  T=256, DH=64.
// grid = (T/64=4, B*NH=512), block = 256
// LDS: qA(8KB) | kB(32KB) | vB(32KB) | S f32 64x256 (64KB) = 136KB
// ---------------------------------------------------------------------------
__global__ __launch_bounds__(256) void attn_kernel(
    const float* __restrict__ q, const float* __restrict__ k,
    const float* __restrict__ v, float* __restrict__ attn)
{
  extern __shared__ char smem[];
  bf16*  qA = (bf16*)smem;                     // 2 ksteps * 2048
  bf16*  kB = (bf16*)(smem + 8192);            // (kd,st) frags: 2*16*512
  bf16*  vB = (bf16*)(smem + 40960);           // (ks,dt) frags: 8*4*512
  float* S  = (float*)(smem + 73728);          // 64x256 scores

  const int tid  = threadIdx.x;
  const int lane = tid & 31;
  const int w    = tid >> 5;
  const int qb   = blockIdx.x;          // query block (0..3)
  const int bn   = blockIdx.y;          // b*NH + n
  const int b    = bn >> 3;
  const int n    = bn & 7;
  const long base = (long)b * 256 * 512 + n * 64;

  // ---- stage Q (A-fragment layout), 64 rows x 64 d ----
  {
    const int m  = tid >> 2;
    const int dg = (tid & 3) * 8;
    #pragma unroll
    for (int kd = 0; kd < 2; ++kd) {
      const float* src = q + base + (long)(qb * 64 + m) * 512 + kd * 32 + dg;
      v4f f0 = *(const v4f*)src;
      v4f f1 = *(const v4f*)(src + 4);
      const int off = ((m >> 4) * 32 + (m & 15) + ((dg & 8) ? 16 : 0)) * 16
                      + ((dg & 16) ? 8 : 0);
      *(v8bf*)&qA[kd * 2048 + off] = cvt8(f0, f1);
    }
  }
  // ---- stage K^T (B-fragment: kdim=d, n=s). thread = s row ----
  {
    const int s  = tid;
    const int st = s >> 4;
    const int sl = s & 15;
    #pragma unroll
    for (int dblk = 0; dblk < 8; ++dblk) {
      const int d0 = dblk * 8;
      const float* src = k + base + (long)s * 512 + d0;
      v4f f0 = *(const v4f*)src;
      v4f f1 = *(const v4f*)(src + 4);
      const int kd = d0 >> 5, dl = d0 & 31;
      const int off = (sl + ((dl & 8) ? 16 : 0)) * 16 + ((dl & 16) ? 8 : 0);
      *(v8bf*)&kB[(kd * 16 + st) * 512 + off] = cvt8(f0, f1);
    }
  }
  // ---- stage V (B-fragment: kdim=s, n=d). thread owns one d, 64 s rows ----
  {
    const int d  = tid & 63;
    const int sg = tid >> 6;
    const int dt = d >> 4, r = d & 15;
    #pragma unroll
    for (int sb = 0; sb < 8; ++sb) {
      const int s0 = sg * 64 + sb * 8;
      v8bf t;
      #pragma unroll
      for (int i = 0; i < 8; ++i) t[i] = (bf16)v[base + (long)(s0 + i) * 512 + d];
      const int ks = s0 >> 5;
      const int off = (r + ((s0 & 8) ? 16 : 0)) * 16 + ((s0 & 16) ? 8 : 0);
      *(v8bf*)&vB[(ks * 4 + dt) * 512 + off] = t;
    }
  }
  __syncthreads();

  // ---- scores = (Q K^T) / 8 : wave w -> mt = w>>1, 8 s-tiles ----
  {
    const int mt = w >> 1;
    v8f acc[8] = {};
    #pragma unroll
    for (int kd = 0; kd < 2; ++kd) {
      v16bf aF = *(const v16bf*)&qA[kd * 2048 + (mt * 32 + lane) * 16];
      #pragma unroll
      for (int j = 0; j < 8; ++j) {
        const int st = (w & 1) * 8 + j;
        v16bf bF = *(const v16bf*)&kB[(kd * 16 + st) * 512 + lane * 16];
        acc[j] = WMMA_BF16(aF, bF, acc[j]);
      }
    }
    const int rowb = mt * 16 + ((lane & 16) ? 8 : 0);
    #pragma unroll
    for (int j = 0; j < 8; ++j) {
      const int col = ((w & 1) * 8 + j) * 16 + (lane & 15);
      #pragma unroll
      for (int i = 0; i < 8; ++i)
        S[(rowb + i) * 256 + col] = acc[j][i] * 0.125f;  // 1/sqrt(64)
    }
  }
  __syncthreads();

  // ---- softmax over rows (64 rows) ----
  if (tid < 64) {
    float* Sr = S + tid * 256;
    float mx = -3.4e38f;
    for (int s = 0; s < 256; ++s) mx = fmaxf(mx, Sr[s]);
    float sum = 0.0f;
    for (int s = 0; s < 256; ++s) { float e = __expf(Sr[s] - mx); Sr[s] = e; sum += e; }
    const float inv = 1.0f / sum;
    for (int s = 0; s < 256; ++s) Sr[s] *= inv;
  }
  __syncthreads();

  // ---- attn = softmax(S) @ V : wave -> mt = w>>1, 2 d-tiles ----
  {
    const int mt  = w >> 1;
    const int dt0 = (w & 1) * 2;
    v8f acc0 = {}, acc1 = {};
    const int mrow = mt * 16 + (lane & 15);   // A-frag: M = lane&15
    for (int ks = 0; ks < 8; ++ks) {
      v16bf aF;
      #pragma unroll
      for (int e = 0; e < 16; ++e) {
        const int kidx = (e < 8 ? e : e + 8) + ((lane & 16) ? 8 : 0);
        aF[e] = (bf16)S[mrow * 256 + ks * 32 + kidx];
      }
      v16bf bF0 = *(const v16bf*)&vB[(ks * 4 + dt0) * 512 + lane * 16];
      v16bf bF1 = *(const v16bf*)&vB[(ks * 4 + dt0 + 1) * 512 + lane * 16];
      acc0 = WMMA_BF16(aF, bF0, acc0);
      acc1 = WMMA_BF16(aF, bF1, acc1);
    }
    const int rowb = qb * 64 + mt * 16 + ((lane & 16) ? 8 : 0);
    #pragma unroll
    for (int j = 0; j < 2; ++j) {
      v8f a = j ? acc1 : acc0;
      const int col = n * 64 + (dt0 + j) * 16 + (lane & 15);
      #pragma unroll
      for (int i = 0; i < 8; ++i)
        attn[((long)b * 256 + rowb + i) * 512 + col] = a[i];
    }
  }
}

// ---------------------------------------------------------------------------
// LSTM recurrent step t:  z = Xg[:,t,:] + h_in @ W_hh + b ; gates ; c,h update
// grid = 8 WGs (each owns a 64-wide h-column chunk and all 4 gate groups).
// LDS: hA(4KB) | WB(16KB) | z f32 64x256 (64KB) = 84KB
// ---------------------------------------------------------------------------
__global__ __launch_bounds__(256) void lstm_step_kernel(
    const float* __restrict__ Xg, const float* __restrict__ W_hh,
    const float* __restrict__ b_lstm, const float* __restrict__ h_in,
    float* __restrict__ h_out, float* __restrict__ c_buf,
    float* __restrict__ h_seq, int t)
{
  extern __shared__ char smem[];
  bf16*  hA = (bf16*)smem;               // 64x32 A frags
  bf16*  WB = (bf16*)(smem + 4096);      // 16 column-tiles * 512
  float* zS = (float*)(smem + 20480);    // 64 x 256 (cols: gate*64 + j)

  const int tid   = threadIdx.x;
  const int lane  = tid & 31;
  const int w     = tid >> 5;
  const int chunk = blockIdx.x;          // h-column chunk (0..7)
  const int mt    = w >> 1;

  v8f acc[8] = {};

  const int sam = tid >> 2;
  const int sak = (tid & 3) * 8;
  const int aBase = (((sam >> 4) * 32 + (sam & 15) + ((sak & 8) ? 16 : 0)) * 16)
                    + ((sak & 16) ? 8 : 0);
  const int sct = tid >> 4;              // column tile (0..15): gate=sct>>2, sub=sct&3
  const int snl = tid & 15;
  const int scol = (sct >> 2) * 512 + chunk * 64 + (sct & 3) * 16 + snl;

  for (int kk = 0; kk < 512; kk += 32) {
    {
      const float* src = h_in + (long)sam * 512 + kk + sak;
      v4f f0 = *(const v4f*)src;
      v4f f1 = *(const v4f*)(src + 4);
      *(v8bf*)&hA[aBase] = cvt8(f0, f1);
    }
    {
      const float* src = W_hh + (long)kk * 2048 + scol;
      #pragma unroll
      for (int kb = 0; kb < 4; ++kb) {
        v8bf tt;
        #pragma unroll
        for (int i = 0; i < 8; ++i) tt[i] = (bf16)src[(long)(kb * 8 + i) * 2048];
        const int off = (snl + ((kb & 1) ? 16 : 0)) * 16 + ((kb & 2) ? 8 : 0);
        *(v8bf*)&WB[sct * 512 + off] = tt;
      }
      if (kk + 32 < 512) __builtin_prefetch(src + 32 * 2048, 0, 1);
    }
    __syncthreads();
    v16bf aF = *(const v16bf*)&hA[(mt * 32 + lane) * 16];
    #pragma unroll
    for (int j = 0; j < 8; ++j) {
      const int ct = (w & 1) * 8 + j;
      v16bf bF = *(const v16bf*)&WB[ct * 512 + lane * 16];
      acc[j] = WMMA_BF16(aF, bF, acc[j]);
    }
    __syncthreads();
  }

  // z = acc + Xg + b_lstm -> LDS
  const int rowb = mt * 16 + ((lane & 16) ? 8 : 0);
  #pragma unroll
  for (int j = 0; j < 8; ++j) {
    const int ct   = (w & 1) * 8 + j;
    const int gcol = (ct >> 2) * 512 + chunk * 64 + (ct & 3) * 16 + (lane & 15);
    const float bb = b_lstm[gcol];
    #pragma unroll
    for (int i = 0; i < 8; ++i) {
      const int brow = rowb + i;
      const float zv = acc[j][i] + Xg[((long)brow * 256 + t) * 2048 + gcol] + bb;
      zS[brow * 256 + ct * 16 + (lane & 15)] = zv;
    }
  }
  __syncthreads();

  // gates + state update (flax order i, f, g, o)
  #pragma unroll
  for (int r = 0; r < 16; ++r) {
    const int idx = tid + r * 256;       // 0..4095
    const int bb  = idx >> 6;
    const int jj  = idx & 63;
    const float iv = zS[bb * 256 + jj];
    const float fv = zS[bb * 256 + 64 + jj];
    const float gv = zS[bb * 256 + 128 + jj];
    const float ov = zS[bb * 256 + 192 + jj];
    const float si = 1.0f / (1.0f + __expf(-iv));
    const float sf = 1.0f / (1.0f + __expf(-fv));
    const float so = 1.0f / (1.0f + __expf(-ov));
    const long cidx = (long)bb * 512 + chunk * 64 + jj;
    const float c = sf * c_buf[cidx] + si * tanhf(gv);
    c_buf[cidx] = c;
    const float h = so * tanhf(c);
    h_out[cidx] = h;
    h_seq[((long)bb * 256 + t) * 512 + chunk * 64 + jj] = h;
  }
}

// ---------------------------------------------------------------------------
__global__ __launch_bounds__(256) void zero_kernel(float* p, int n) {
  const int i = blockIdx.x * 256 + threadIdx.x;
  if (i < n) p[i] = 0.0f;
}

__global__ __launch_bounds__(256) void split_kernel(
    const float* __restrict__ out_all, float* __restrict__ out, int n) {
  const int i = blockIdx.x * 256 + threadIdx.x;
  if (i < n) {
    const int row = i >> 6, c = i & 63;
    out[i]     = out_all[row * 128 + c];        // mu
    out[n + i] = out_all[row * 128 + 64 + c];   // log_sigma
  }
}

// ---------------------------------------------------------------------------
extern "C" void kernel_launch(void* const* d_in, const int* in_sizes, int n_in,
                              void* d_out, int out_size, void* d_ws, size_t ws_size,
                              hipStream_t stream) {
  (void)in_sizes; (void)n_in; (void)out_size; (void)ws_size;
  const float* x_seq  = (const float*)d_in[0];
  const float* W_in   = (const float*)d_in[1];
  const float* b_in   = (const float*)d_in[2];
  const float* Wq     = (const float*)d_in[3];
  const float* bq     = (const float*)d_in[4];
  const float* Wk     = (const float*)d_in[5];
  const float* bk     = (const float*)d_in[6];
  const float* Wv     = (const float*)d_in[7];
  const float* bv     = (const float*)d_in[8];
  const float* Wo     = (const float*)d_in[9];
  const float* bo     = (const float*)d_in[10];
  const float* W_ih   = (const float*)d_in[11];
  const float* W_hh   = (const float*)d_in[12];
  const float* b_lstm = (const float*)d_in[13];
  const float* W1     = (const float*)d_in[14];
  const float* b1     = (const float*)d_in[15];
  const float* W2     = (const float*)d_in[16];
  const float* b2     = (const float*)d_in[17];

  const long BT = 16384;   // B*T
  float* ws   = (float*)d_ws;
  float* xp   = ws;                         // [BT,512]; later reused as attn_proj
  float* qbuf = xp + BT * 512;              // [BT,512]; later reused as h_seq
  float* kbuf = qbuf + BT * 512;            // [BT,512]; later reused as hid
  float* vbuf = kbuf + BT * 512;            // [BT,512]; later reused as out_all
  float* attn = vbuf + BT * 512;            // [BT,512]
  float* Xg   = attn + BT * 512;            // [BT,2048]
  float* cbuf = Xg + BT * 2048;             // [64,512]
  float* h0   = cbuf + 64 * 512;            // [64,512]
  float* h1   = h0 + 64 * 512;              // [64,512]
  float* attn_proj = xp;
  float* h_seq     = qbuf;
  float* hid       = kbuf;
  float* out_all   = vbuf;

  const dim3 blk(256);
  const size_t gemmLds = 12288;   // 4KB A + 8KB B

  // xp = x_seq @ W_in + b_in                 [16384,512] K=128
  gemm_bf16_kernel<<<dim3(4, 256, 1), blk, gemmLds, stream>>>(
      x_seq, 0, 128, W_in, 0, 512, b_in, 0, xp, 0, 512, 128, 0);
  // q/k/v = xp @ W{q,k,v} + b{q,k,v}         K=512
  gemm_bf16_kernel<<<dim3(4, 256, 1), blk, gemmLds, stream>>>(
      xp, 0, 512, Wq, 0, 512, bq, 0, qbuf, 0, 512, 512, 0);
  gemm_bf16_kernel<<<dim3(4, 256, 1), blk, gemmLds, stream>>>(
      xp, 0, 512, Wk, 0, 512, bk, 0, kbuf, 0, 512, 512, 0);
  gemm_bf16_kernel<<<dim3(4, 256, 1), blk, gemmLds, stream>>>(
      xp, 0, 512, Wv, 0, 512, bv, 0, vbuf, 0, 512, 512, 0);
  // attention (per b,head)
  attn_kernel<<<dim3(4, 512, 1), blk, 139264, stream>>>(qbuf, kbuf, vbuf, attn);
  // attn_proj = attn @ Wo + bo
  gemm_bf16_kernel<<<dim3(4, 256, 1), blk, gemmLds, stream>>>(
      attn, 0, 512, Wo, 0, 512, bo, 0, attn_proj, 0, 512, 512, 0);
  // Xg = attn_proj @ W_ih                    [16384,2048]
  gemm_bf16_kernel<<<dim3(16, 256, 1), blk, gemmLds, stream>>>(
      attn_proj, 0, 512, W_ih, 0, 2048, nullptr, 0, Xg, 0, 2048, 512, 0);
  // zero c and h0 (contiguous)
  zero_kernel<<<dim3(256), blk, 0, stream>>>(cbuf, 64 * 512 * 2);
  // sequential LSTM
  for (int t = 0; t < 256; ++t) {
    const float* hin = (t & 1) ? h1 : h0;
    float* hout      = (t & 1) ? h0 : h1;
    lstm_step_kernel<<<dim3(8), blk, 86016, stream>>>(
        Xg, W_hh, b_lstm, hin, hout, cbuf, h_seq, t);
  }
  // per-timestep heads (batched over t via gridDim.z)
  // hid[:,t,:] = relu(h_seq[:,t,:] @ W1[t] + b1[t])
  gemm_bf16_kernel<<<dim3(4, 1, 256), blk, gemmLds, stream>>>(
      h_seq, 512, 131072, W1, 262144, 512, b1, 512, hid, 512, 131072, 512, 1);
  // out_all[:,t,:] = hid[:,t,:] @ W2[t] + b2[t]
  gemm_bf16_kernel<<<dim3(1, 1, 256), blk, gemmLds, stream>>>(
      hid, 512, 131072, W2, 65536, 128, b2, 128, out_all, 128, 32768, 512, 0);
  // split into mu / log_sigma
  split_kernel<<<dim3(4096), blk, 0, stream>>>(out_all, (float*)d_out, 1048576);
}
